// PruneLayer_73882027426497
// MI455X (gfx1250) — matrix operand order repaired and checked by reading, not compiled, not run on hardware
//
#include <hip/hip_runtime.h>
#include <hip/hip_bf16.h>

// ---------------------------------------------------------------------------
// PruneLayer on MI455X (gfx1250, wave32, WMMA)
//
// Both convs are GEMMs with M=16 (one WMMA tile of output channels),
// K=2304, N=100352, computed with v_wmma_f32_16x16x32_bf16.
// Channel expansion (labels + BN + ReLU) is fused into the LDS slab build of
// the second conv; the 256-channel intermediate never touches HBM.
//
// Tiling: one workgroup = 2 output rows = 112 pixels = 7 N-tiles = 7 waves.
// Tiles are 16 consecutive flattened pixels (may cross a row boundary); each
// lane computes its own slab address, so no WMMA lane is ever padding.
// ---------------------------------------------------------------------------

typedef __bf16 bf16;
typedef __attribute__((ext_vector_type(16))) __bf16 v16bf;
typedef __attribute__((ext_vector_type(8)))  float  v8f;

#define BB 32
#define CC 256
#define HH 56
#define WW 56
#define HW (HH*WW)
#define KSTEPS 72           // 9 taps * 8 blocks of 32 channels = K 2304
#define FRAG_ELEMS 512      // one 16x32 bf16 A-fragment per K-step
#define ROWSPB 2            // output rows per workgroup
#define SLABROWS 4          // input rows staged: h0-1 .. h0+2
#define SLABCOLS 58         // input cols -1 .. 56, zero halo at both ends
#define NTHREADS 224        // 7 waves; 1 slab position per thread

// ---------------------------------------------------------------------------
// Kernel 0: fold BN/bias/labels into per-channel (alpha, delta, lab) and
// pre-swizzle w1/w2 (f32 [16,256,3,3]) into bf16 A-fragments laid out per the
// contiguous K order matching the B fragment: lane = M (both halves),
// element e -> K = e (+16 for lanes>=16).
// K-step s = r*8 + cb, r = dy*3+dx, channels c = cb*32 + K.
// ---------------------------------------------------------------------------
__global__ void prune_prep_kernel(
    const float* __restrict__ w1, const float* __restrict__ w2,
    const float* __restrict__ b1, const float* __restrict__ b2,
    const float* __restrict__ g1, const float* __restrict__ be1,
    const float* __restrict__ m1, const float* __restrict__ v1,
    const float* __restrict__ g2, const float* __restrict__ be2,
    const float* __restrict__ m2, const float* __restrict__ v2,
    const int* __restrict__ l1, const int* __restrict__ l2,
    bf16* __restrict__ fw1, bf16* __restrict__ fw2,
    float* __restrict__ alpha1, float* __restrict__ delta1, int* __restrict__ lab1,
    float* __restrict__ alpha2, float* __restrict__ delta2, int* __restrict__ lab2)
{
    const int tid = threadIdx.x;
    const int total = KSTEPS * FRAG_ELEMS;
    for (int o = tid; o < total; o += blockDim.x) {
        int s   = o >> 9;           // K-step
        int rem = o & 511;
        int l   = rem >> 4;         // lane
        int e   = rem & 15;         // element within v16bf
        int r   = s >> 3;           // tap index (dy*3+dx)
        int cb  = s & 7;            // 32-channel block
        int dy  = r / 3, dx = r % 3;
        int m   = l & 15;
        int hi  = (l >= 16) ? 16 : 0;
        int K   = e + hi;
        int c   = cb*32 + K;
        int wi  = ((m*CC + c)*3 + dy)*3 + dx;
        fw1[o] = (bf16)w1[wi];
        fw2[o] = (bf16)w2[wi];
    }
    if (tid < CC) {
        int c = tid;
        float i1 = g1[c] * rsqrtf(v1[c] + 1e-5f);
        int  L1 = l1[c];
        alpha1[c] = i1;
        delta1[c] = i1*b1[L1] + be1[c] - m1[c]*i1;
        lab1[c]   = L1;
        float i2 = g2[c] * rsqrtf(v2[c] + 1e-5f);
        int  L2 = l2[c];
        alpha2[c] = i2;
        delta2[c] = i2*b2[L2] + be2[c] - m2[c]*i2;
        lab2[c]   = L2;
    }
}

// ---------------------------------------------------------------------------
// Kernels 1/2: implicit-GEMM 3x3 conv, 16 output channels.
//   stage 1: src = x [B,256,H,W]                  -> dst = z1 [16,B,H,W]
//   stage 2: src = z1 [16,B,H,W], expand channels -> dst = z2 [16,B,H,W]
//            h[c] = relu(alpha[c]*z1[lab[c]] + delta[c])  (built in LDS)
// LDS slab: channel-last bf16 [4 rows][58 cols][32 ch], zero halo.
// Fill: one (row,col) per thread; 32 immediate-offset b32 loads -> 4x b128 ds.
// B fragment: one 32B aligned LDS read per lane at base + (dy*58+dx)*64 bytes
// (immediate ds offset); lanes0-15 hold K=0..15, lanes16-31 K=16..31.
// ---------------------------------------------------------------------------
__global__ void __launch_bounds__(NTHREADS)
prune_conv_wmma_kernel(const float* __restrict__ src,
                       const bf16*  __restrict__ wfrag,
                       const float* __restrict__ alpha,
                       const float* __restrict__ delta,
                       const int*   __restrict__ lab,
                       float* __restrict__ dst,
                       int stage)
{
    __shared__ bf16 xs[SLABROWS * SLABCOLS * 32];   // 14,848 B

    const int tid  = threadIdx.x;
    const int lane = tid & 31;
    const int wave = tid >> 5;                      // 0..6 = N-tile index
    const int blk  = blockIdx.x;
    const int b    = blk / (HH / ROWSPB);
    const int h0   = (blk % (HH / ROWSPB)) * ROWSPB;

    // this lane's output pixel (flattened over the 2-row block)
    const int n      = lane & 15;                   // N within tile
    const int pix    = wave * 16 + n;               // 0..111
    const int h_lane = h0 + pix / WW;
    const int w_lane = pix % WW;
    const int koff   = (lane >= 16) ? 16 : 0;       // B-fragment K half
    // slab base element for tap (dy=0,dx=0): row (h_lane-h0), col w_lane
    const int bbase  = ((h_lane - h0) * SLABCOLS + w_lane) * 32 + koff;

    // this thread's slab fill position (exactly one per thread)
    const int frow = tid / WW;                      // 0..3
    const int fcol = tid % WW;                      // 0..55
    const int gh   = h0 - 1 + frow;                 // -1 .. 56
    const bool gok = (gh >= 0 && gh < HH);
    bf16* fdst = &xs[(frow * SLABCOLS + (fcol + 1)) * 32];

    // zero the slab once (b128 stores); halo cols / OOB rows stay zero
    {
        const uint4 zq = make_uint4(0u, 0u, 0u, 0u);
        for (int i = tid * 8; i < SLABROWS*SLABCOLS*32; i += NTHREADS * 8)
            *(uint4*)&xs[i] = zq;
    }

    v8f acc = {};

    for (int cb = 0; cb < 8; ++cb) {
        __syncthreads();
        if (gok) {
            union { bf16 e[32]; uint4 q[4]; } buf;
            if (stage == 1) {
                const float* sp = src + (((long)b*CC + cb*32)*HH + gh)*WW + fcol;
                if (cb < 7)  // hide next channel block behind the WMMA loop
                    __builtin_prefetch(sp + 32*(long)HW, 0, 1);
#pragma unroll
                for (int c = 0; c < 32; ++c)
                    buf.e[c] = (bf16)sp[(long)c * HW];
            } else {
#pragma unroll
                for (int c = 0; c < 32; ++c) {
                    int   C = cb*32 + c;                   // wave-uniform index
                    float z = src[(((long)lab[C]*BB + b)*HH + gh)*WW + fcol];
                    float val = alpha[C]*z + delta[C];
                    buf.e[c] = (bf16)(val > 0.0f ? val : 0.0f);
                }
            }
            uint4* dp = (uint4*)fdst;
            dp[0] = buf.q[0]; dp[1] = buf.q[1];
            dp[2] = buf.q[2]; dp[3] = buf.q[3];
        }
        __syncthreads();

#pragma unroll
        for (int r = 0; r < 9; ++r) {
            const int dy = r / 3, dx = r % 3;
            // A fragment: 32B per lane from pre-swizzled weights (L2 broadcast)
            v16bf a = *(const v16bf*)(wfrag + (r*8 + cb)*FRAG_ELEMS + lane*16);
            // B fragment: 32B aligned LDS read, immediate offset per tap
            v16bf bfrag = *(const v16bf*)&xs[bbase + (dy*SLABCOLS + dx)*32];
            acc = __builtin_amdgcn_wmma_f32_16x16x32_bf16(
                      false, a, false, bfrag, (short)0, acc, false, false);
        }
    }

    // C/D layout: VGPR v, lanes0-15 -> M=v, lanes16-31 -> M=v+8; N = lane&15
    const int mofs = (lane >= 16) ? 8 : 0;
#pragma unroll
    for (int v = 0; v < 8; ++v) {
        dst[((((long)(v + mofs))*BB + b)*HH + h_lane)*WW + w_lane] = acc[v];
    }
}

// ---------------------------------------------------------------------------
// Kernel 3: out = relu(alpha2[c]*z2[lab2[c]] + delta2[c] + x), float4 per thread
// ---------------------------------------------------------------------------
__global__ void prune_epilogue_kernel(const float* __restrict__ x,
                                      const float* __restrict__ z2,
                                      const float* __restrict__ alpha2,
                                      const float* __restrict__ delta2,
                                      const int*   __restrict__ lab2,
                                      float* __restrict__ out)
{
    const long total4 = (long)BB*CC*HW/4;
    long idx = (long)blockIdx.x * blockDim.x + threadIdx.x;
    if (idx >= total4) return;
    long e  = idx * 4;
    int  hw = (int)(e % HW);
    int  c  = (int)((e / HW) % CC);
    int  b  = (int)(e / ((long)HW * CC));
    float a = alpha2[c], d = delta2[c];
    int   L = lab2[c];
    float4 xv = *(const float4*)(x  + e);
    float4 zv = *(const float4*)(z2 + ((long)(L*BB + b)*HW + hw));
    float4 o;
    o.x = fmaxf(a*zv.x + d + xv.x, 0.0f);
    o.y = fmaxf(a*zv.y + d + xv.y, 0.0f);
    o.z = fmaxf(a*zv.z + d + xv.z, 0.0f);
    o.w = fmaxf(a*zv.w + d + xv.w, 0.0f);
    *(float4*)(out + e) = o;
}

// ---------------------------------------------------------------------------
extern "C" void kernel_launch(void* const* d_in, const int* in_sizes, int n_in,
                              void* d_out, int out_size, void* d_ws, size_t ws_size,
                              hipStream_t stream) {
    (void)in_sizes; (void)n_in; (void)out_size; (void)ws_size;

    const float* x   = (const float*)d_in[0];
    const float* w1  = (const float*)d_in[1];
    const float* b1  = (const float*)d_in[2];
    const float* w2  = (const float*)d_in[3];
    const float* b2  = (const float*)d_in[4];
    const float* g1  = (const float*)d_in[5];
    const float* be1 = (const float*)d_in[6];
    const float* m1  = (const float*)d_in[7];
    const float* v1  = (const float*)d_in[8];
    const float* g2  = (const float*)d_in[9];
    const float* be2 = (const float*)d_in[10];
    const float* m2  = (const float*)d_in[11];
    const float* v2  = (const float*)d_in[12];
    const int*   l1  = (const int*)d_in[13];
    const int*   l2  = (const int*)d_in[14];
    float* out = (float*)d_out;

    // workspace layout (~13 MB total)
    char* base = (char*)d_ws;
    const size_t FRAG_BYTES = (size_t)KSTEPS * FRAG_ELEMS * sizeof(bf16); // 73,728
    bf16*  fw1    = (bf16*)(base);
    bf16*  fw2    = (bf16*)(base + FRAG_BYTES);
    char*  pbase  = base + 2*FRAG_BYTES;          // 147,456
    float* alpha1 = (float*)(pbase + 0*1024);
    float* delta1 = (float*)(pbase + 1*1024);
    int*   lab1   = (int*)  (pbase + 2*1024);
    float* alpha2 = (float*)(pbase + 3*1024);
    float* delta2 = (float*)(pbase + 4*1024);
    int*   lab2   = (int*)  (pbase + 5*1024);
    const size_t Z_BYTES = (size_t)16*BB*HW*sizeof(float);   // 6,422,528
    float* z1 = (float*)(pbase + 6*1024);
    float* z2 = (float*)((char*)z1 + Z_BYTES);

    prune_prep_kernel<<<1, 256, 0, stream>>>(
        w1, w2, b1, b2, g1, be1, m1, v1, g2, be2, m2, v2, l1, l2,
        fw1, fw2, alpha1, delta1, lab1, alpha2, delta2, lab2);

    dim3 cgrid(BB * (HH / ROWSPB));   // 896 workgroups, 7 waves each
    prune_conv_wmma_kernel<<<cgrid, NTHREADS, 0, stream>>>(
        x, fw1, alpha1, delta1, lab1, z1, 1);
    prune_conv_wmma_kernel<<<cgrid, NTHREADS, 0, stream>>>(
        z1, fw2, alpha1, delta1, lab1, z2, 2);

    const long total4 = (long)BB*CC*HW/4;
    int eblocks = (int)((total4 + 255) / 256);
    prune_epilogue_kernel<<<eblocks, 256, 0, stream>>>(
        x, z2, alpha2, delta2, lab2, out);
}
